// ScaledDotProductAttention_33560874451491
// MI455X (gfx1250) — matrix operand order repaired and checked by reading, not compiled, not run on hardware
//
#include <hip/hip_runtime.h>
#include <hip/hip_bf16.h>

// ============================================================================
// ScaledDotProductAttention for MI455X (gfx1250, wave32, WMMA bf16 + TDM).
//
//   1) WQ = rnn_ht @ W          (f32 in -> bf16 out, ws)
//   2) WK = Lt @ kernel[0]      (f32 in -> bf16 out, ws)
//   3) WV = Lt @ kernel[1]      (f32 in -> bf16 out, ws)
//   4) QK = WQ @ WK^T * 0.125   (bf16 in via TDM -> f32 out, weights region)
//   5) softmax rows (in place on weights)
//   6) ctx = weights @ WV       (f32+bf16 in -> f32 out, context region)
//
// All GEMM dims are multiples of 128 -> no edge guards.
// Workspace: 3 * 16384*1024 bf16 = 96 MB.
// ============================================================================

typedef __attribute__((ext_vector_type(16))) __bf16        v16bf;
typedef __attribute__((ext_vector_type(8)))  float         v8f;
typedef __attribute__((ext_vector_type(8)))  unsigned int  v8u;
typedef __attribute__((ext_vector_type(4)))  unsigned int  u32x4;
typedef __attribute__((ext_vector_type(8)))  int           i32x8;
typedef __attribute__((ext_vector_type(4)))  int           i32x4;

#if defined(__gfx1250__) && __has_builtin(__builtin_amdgcn_tensor_load_to_lds)
#define HAS_TDM 1
#else
#define HAS_TDM 0
#endif

// Cheap f32->bf16 (round-half-up): 1 v_add + high-half extract.
__device__ __forceinline__ unsigned short f32_to_bf16(float x) {
  return (unsigned short)((__float_as_uint(x) + 0x8000u) >> 16);
}
// Pack two f32 into packed bf16x2 with a single v_perm_b32 after the biases.
__device__ __forceinline__ unsigned pack_bf16(float lo, float hi) {
  const unsigned ulo = __float_as_uint(lo) + 0x8000u;
  const unsigned uhi = __float_as_uint(hi) + 0x8000u;
#if __has_builtin(__builtin_amdgcn_perm)
  return __builtin_amdgcn_perm(uhi, ulo, 0x07060302u);  // {uhi[31:16], ulo[31:16]}
#else
  return (uhi & 0xFFFF0000u) | (ulo >> 16);
#endif
}

// A fragment (16x32 bf16, ISA 7.12.2): lane holds row (lane&15); lane-half
// selects K-interleave base kg=(lane>>4)*8; halves [kg,kg+8) and [kg+16,kg+24).
__device__ __forceinline__ v16bf ld_fragA(const unsigned short* p) {
  const uint4 lo = *(const uint4*)(p);
  const uint4 hi = *(const uint4*)(p + 16);
  v8u raw = {lo.x, lo.y, lo.z, lo.w, hi.x, hi.y, hi.z, hi.w};
  return __builtin_bit_cast(v16bf, raw);
}
// B fragment (32x16 bf16): lane holds column (lane&15); lane-half selects
// K-interleave base kg=(lane>>4)*8; halves [kg,kg+8) and [kg+16,kg+24).
__device__ __forceinline__ v16bf ld_fragB(const unsigned short* p) {
  const uint4 lo = *(const uint4*)(p);
  const uint4 hi = *(const uint4*)(p + 16);
  v8u raw = {lo.x, lo.y, lo.z, lo.w, hi.x, hi.y, hi.z, hi.w};
  return __builtin_bit_cast(v16bf, raw);
}

__device__ __forceinline__ void wait_tensorcnt0() {
#if __has_builtin(__builtin_amdgcn_s_wait_tensorcnt)
  __builtin_amdgcn_s_wait_tensorcnt(0);
#elif defined(__gfx1250__)
  asm volatile("s_wait_tensorcnt 0x0" ::: "memory");
#endif
}

#define LDT 48        // padded LDS row stride in halves (96B, 16B-aligned)
#define TILE_BYTES (128 * LDT * 2)  // 12288 B per tile

#if HAS_TDM
// One TDM descriptor load: 128 rows x 32 bf16 tile, row stride ld elements,
// into LDS at byte offset lds_off with 8-DWORD pad after every 16 DWORDs
// (=> 48-half LDS row stride). ISA 8.3/8.4 field layout.
// This toolchain exposes the 6-arg builtin:
//   (u32x4 g0, i32x8 g1, i32x4 g2, i32x4 g3, i32x8 extra, i32 cpol)
__device__ __forceinline__ void tdm_tile_load(const void* gsrc, unsigned lds_off,
                                              unsigned ld) {
  const unsigned long long ga = (unsigned long long)gsrc;
  u32x4 g0 = {
      1u,                                                   // count=1 (valid D#)
      lds_off,                                              // lds_addr [63:32]
      (unsigned)ga,                                         // global_addr lo
      ((unsigned)(ga >> 32) & 0x01FFFFFFu) | (2u << 30)     // addr[56:32] | type=2
  };
  i32x8 g1 = {
      (int)((1u << 16)        // data_size = 2 bytes
            | (1u << 20)      // pad_enable
            | (3u << 22)      // pad_interval: 16 DWORDs (64B) before padding
            | (7u << 25)),    // pad_amount: 8 DWORDs (32B)
      (int)(32u << 16),       // [63:48]=tensor_dim0 lo16 = 32 (barrier addr = 0)
      (int)(128u << 16),      // [95:80]=tensor_dim1 lo16 = 128 (dim0 hi16 = 0)
      (int)(32u << 16),       // [127:112]=tile_dim0 = 32 (dim1 hi16 = 0)
      (int)128,               // [143:128]=tile_dim1 = 128
      (int)ld,                // [191:160]=tensor_dim0_stride lo32
      0, 0                    // stride hi / dim1_stride (unused for 2D tile)
  };
  i32x4 gz4 = {0, 0, 0, 0};          // groups 2/3 unused (2D tensor)
  i32x8 gz8 = {0, 0, 0, 0, 0, 0, 0, 0};
  __builtin_amdgcn_tensor_load_to_lds(g0, g1, gz4, gz4, gz8, 0);
}
#endif

// AMODE: 0 = A is f32 [M,K] (convert), 1 = A is bf16 [M,K] (copy)
// BMODE: 0 = B is bf16 [N,K] (copy, for A@B^T), 1 = B is f32 [K,N] (transpose+cvt),
//        2 = B is bf16 [K,N] (transpose)
// OUTMODE: 0 = bf16 out, 1 = f32 out
// USETDM: 1 = stage tiles with tensor_load_to_lds (requires AMODE=1, BMODE=0)
template<int AMODE, int BMODE, int OUTMODE, int USETDM>
__global__ __launch_bounds__(256)
void gemm_wmma_kernel(const void* __restrict__ Ap, const void* __restrict__ Bp,
                      void* __restrict__ Cp, int K, int lda, int ldb, int ldc,
                      long long strideA, long long strideB, long long strideC,
                      float scale) {
  __shared__ __align__(16) unsigned short lds[2][2][128][LDT];  // [buf][A/B][row][k]

  const int tid  = threadIdx.x;
  const int lane = tid & 31;
  const int wave = tid >> 5;
  const int waveM = wave >> 1;             // 0..3  (32-row slab)
  const int waveN = wave & 1;              // 0..1  (64-col slab)
  const int row0 = blockIdx.y * 128;
  const int col0 = blockIdx.x * 128;
  const long long bz = blockIdx.z;

  const float*          A32 = (const float*)Ap          + bz * strideA;
  const unsigned short* A16 = (const unsigned short*)Ap + bz * strideA;
  const float*          B32 = (const float*)Bp          + bz * strideB;
  const unsigned short* B16 = (const unsigned short*)Bp + bz * strideB;
  float*          C32 = (float*)Cp          + bz * strideC;
  unsigned short* C16 = (unsigned short*)Cp + bz * strideC;

  // Staging maps: 128x32-element tile, 256 threads, 16 elements each.
  const int ar  = tid >> 1;                // row-copy map: 0..127
  const int ah  = tid & 1;                 // k half: 0 -> k[0,16), 1 -> k[16,32)
  const int bk  = tid >> 3;                // transpose map: k-row 0..31
  const int bn0 = (tid & 7) * 16;          // 16 consecutive n per thread

  float4 aregf[4]; uint4 aregh[2];
  float4 bregf[4]; uint4 bregh[2];

  auto loadA = [&](int kt) {
    if (AMODE == 0) {
      const float* p = A32 + (long long)(row0 + ar) * lda + kt * 32 + ah * 16;
      #pragma unroll
      for (int i = 0; i < 4; ++i) aregf[i] = *(const float4*)(p + i * 4);
      __builtin_prefetch(p + 64, 0, 1);                 // global_prefetch_b8
    } else {
      const unsigned short* p = A16 + (long long)(row0 + ar) * lda + kt * 32 + ah * 16;
      aregh[0] = *(const uint4*)(p);
      aregh[1] = *(const uint4*)(p + 8);
      __builtin_prefetch(p + 64, 0, 1);
    }
  };
  auto storeA = [&](int buf) {
    unsigned short* d = &lds[buf][0][ar][ah * 16];
    if (AMODE == 0) {
      uint4 p0, p1;
      p0.x = pack_bf16(aregf[0].x, aregf[0].y); p0.y = pack_bf16(aregf[0].z, aregf[0].w);
      p0.z = pack_bf16(aregf[1].x, aregf[1].y); p0.w = pack_bf16(aregf[1].z, aregf[1].w);
      p1.x = pack_bf16(aregf[2].x, aregf[2].y); p1.y = pack_bf16(aregf[2].z, aregf[2].w);
      p1.z = pack_bf16(aregf[3].x, aregf[3].y); p1.w = pack_bf16(aregf[3].z, aregf[3].w);
      *(uint4*)(d) = p0; *(uint4*)(d + 8) = p1;
    } else {
      *(uint4*)(d) = aregh[0]; *(uint4*)(d + 8) = aregh[1];
    }
  };
  auto loadB = [&](int kt) {
    if (BMODE == 0) {
      const unsigned short* p = B16 + (long long)(col0 + ar) * ldb + kt * 32 + ah * 16;
      bregh[0] = *(const uint4*)(p); bregh[1] = *(const uint4*)(p + 8);
      __builtin_prefetch(p + 64, 0, 1);
    } else if (BMODE == 1) {
      const float* p = B32 + (long long)(kt * 32 + bk) * ldb + col0 + bn0;
      #pragma unroll
      for (int i = 0; i < 4; ++i) bregf[i] = *(const float4*)(p + i * 4);
    } else {
      const unsigned short* p = B16 + (long long)(kt * 32 + bk) * ldb + col0 + bn0;
      bregh[0] = *(const uint4*)(p); bregh[1] = *(const uint4*)(p + 8);
      __builtin_prefetch(p + 64, 0, 1);
    }
  };
  auto storeB = [&](int buf) {
    if (BMODE == 0) {
      unsigned short* d = &lds[buf][1][ar][ah * 16];
      *(uint4*)(d) = bregh[0]; *(uint4*)(d + 8) = bregh[1];
    } else if (BMODE == 1) {
      #pragma unroll
      for (int i = 0; i < 4; ++i) {
        const float* f = (const float*)&bregf[i];
        #pragma unroll
        for (int e = 0; e < 4; ++e)
          lds[buf][1][bn0 + i * 4 + e][bk] = f32_to_bf16(f[e]);
      }
    } else {
      const unsigned short* h = (const unsigned short*)&bregh[0];
      #pragma unroll
      for (int e = 0; e < 16; ++e) lds[buf][1][bn0 + e][bk] = h[e];
    }
  };

  v8f acc[2][4] = {};
  const int fr = lane & 15;
  const int kg = (lane >> 4) * 8;    // A fragment K-interleave base
  const int bh = (lane >> 4) * 16;   // unused: B fragment also uses kg

  auto compute = [&](int buf) {
    v16bf af[2], bfr[4];
    #pragma unroll
    for (int tm = 0; tm < 2; ++tm)
      af[tm] = ld_fragA(&lds[buf][0][waveM * 32 + tm * 16 + fr][kg]);
    #pragma unroll
    for (int tn = 0; tn < 4; ++tn)
      bfr[tn] = ld_fragB(&lds[buf][1][waveN * 64 + tn * 16 + fr][kg]);
    #pragma unroll
    for (int tm = 0; tm < 2; ++tm)
      #pragma unroll
      for (int tn = 0; tn < 4; ++tn)
        acc[tm][tn] = __builtin_amdgcn_wmma_f32_16x16x32_bf16(
            false, af[tm], false, bfr[tn], (short)0, acc[tm][tn], false, false);
  };

  // C/D layout (ISA 7.12.2): lane holds column (lane&15); VGPR v holds row
  // v + 8*(lane>>4).
  auto writeC = [&]() {
    const int hi8 = (lane >> 4) * 8;
    #pragma unroll
    for (int tm = 0; tm < 2; ++tm) {
      #pragma unroll
      for (int tn = 0; tn < 4; ++tn) {
        const int m0 = row0 + waveM * 32 + tm * 16 + hi8;
        const int n  = col0 + waveN * 64 + tn * 16 + fr;
        #pragma unroll
        for (int v = 0; v < 8; ++v) {
          const float val = acc[tm][tn][v] * scale;
          if (OUTMODE == 1) C32[(long long)(m0 + v) * ldc + n] = val;
          else              C16[(long long)(m0 + v) * ldc + n] = f32_to_bf16(val);
        }
      }
    }
  };

  const int KT = K >> 5;

#if HAS_TDM
  if (USETDM) {
    // Static LDS block starts at offset 0 of the wave's LDS allocation.
    // Wave 0 drives the Tensor Data Mover; all waves consume via WMMA.
    if (wave == 0) {
      tdm_tile_load(A16 + (long long)row0 * lda, 0 * TILE_BYTES, lda);
      tdm_tile_load(B16 + (long long)col0 * ldb, 1 * TILE_BYTES, ldb);
      wait_tensorcnt0();
    }
    __syncthreads();
    for (int kt = 0; kt < KT; ++kt) {
      const int cb = kt & 1;
      if (wave == 0 && kt + 1 < KT) {
        const int nb = cb ^ 1;
        tdm_tile_load(A16 + (long long)row0 * lda + (kt + 1) * 32,
                      (nb * 2 + 0) * TILE_BYTES, lda);
        tdm_tile_load(B16 + (long long)col0 * ldb + (kt + 1) * 32,
                      (nb * 2 + 1) * TILE_BYTES, ldb);
      }
      compute(cb);
      if (wave == 0) wait_tensorcnt0();
      __syncthreads();
    }
    writeC();
    return;
  }
#endif

  loadA(0); loadB(0);
  storeA(0); storeB(0);
  __syncthreads();
  for (int kt = 0; kt < KT; ++kt) {
    const int cb = kt & 1;
    if (kt + 1 < KT) { loadA(kt + 1); loadB(kt + 1); }
    compute(cb);
    if (kt + 1 < KT) { storeA(cb ^ 1); storeB(cb ^ 1); }
    __syncthreads();
  }
  writeC();
}

// Row softmax over 2048 columns, one block per row, register-resident.
__global__ __launch_bounds__(256)
void softmax2048_kernel(float* __restrict__ w) {
  __shared__ float red[256];
  float* row = w + (long long)blockIdx.x * 2048;
  const int t = threadIdx.x;
  float v[8];
  #pragma unroll
  for (int i = 0; i < 8; ++i) v[i] = row[t + i * 256];
  float m = v[0];
  #pragma unroll
  for (int i = 1; i < 8; ++i) m = fmaxf(m, v[i]);
  red[t] = m; __syncthreads();
  for (int s = 128; s > 0; s >>= 1) {
    if (t < s) red[t] = fmaxf(red[t], red[t + s]);
    __syncthreads();
  }
  m = red[0]; __syncthreads();
  float sum = 0.f;
  #pragma unroll
  for (int i = 0; i < 8; ++i) { v[i] = __expf(v[i] - m); sum += v[i]; }
  red[t] = sum; __syncthreads();
  for (int s = 128; s > 0; s >>= 1) {
    if (t < s) red[t] += red[t + s];
    __syncthreads();
  }
  const float inv = 1.0f / red[0];
  #pragma unroll
  for (int i = 0; i < 8; ++i) row[t + i * 256] = v[i] * inv;
}

extern "C" void kernel_launch(void* const* d_in, const int* in_sizes, int n_in,
                              void* d_out, int out_size, void* d_ws, size_t ws_size,
                              hipStream_t stream) {
  const float* Lt   = (const float*)d_in[0];   // [8,2048,1024]
  const float* rnn  = (const float*)d_in[1];   // [8,2048,1024]
  const float* kern = (const float*)d_in[2];   // [2,1024,1024]
  const float* W    = (const float*)d_in[3];   // [1024,1024]

  const long long B = 8, SQ = 2048, SK = 2048, D = 1024, E = 1024;

  float* ctx = (float*)d_out;                  // [8,2048,1024]
  float* wts = ctx + B * SQ * E;               // [8,2048,2048]

  unsigned short* wq = (unsigned short*)d_ws;  // bf16 [8*2048,1024]
  unsigned short* wk = wq + B * SQ * E;
  unsigned short* wv = wk + B * SK * E;

  dim3 blk(256, 1, 1);

  // 1-3) Projections: [16384,1024] x [1024,1024] -> bf16
  dim3 gp((unsigned)(E / 128), (unsigned)((B * SQ) / 128), 1);
  gemm_wmma_kernel<0, 1, 0, 0><<<gp, blk, 0, stream>>>(
      rnn, W, wq, (int)D, (int)D, (int)E, (int)E, 0LL, 0LL, 0LL, 1.0f);
  gemm_wmma_kernel<0, 1, 0, 0><<<gp, blk, 0, stream>>>(
      Lt, kern, wk, (int)D, (int)D, (int)E, (int)E, 0LL, 0LL, 0LL, 1.0f);
  gemm_wmma_kernel<0, 1, 0, 0><<<gp, blk, 0, stream>>>(
      Lt, kern + D * E, wv, (int)D, (int)D, (int)E, (int)E, 0LL, 0LL, 0LL, 1.0f);

  // 4) QK^T * 1/sqrt(64): per batch [2048,1024] x [2048,1024]^T -> f32 (TDM path)
  dim3 gq((unsigned)(SK / 128), (unsigned)(SQ / 128), (unsigned)B);
  gemm_wmma_kernel<1, 0, 1, 1><<<gq, blk, 0, stream>>>(
      wq, wk, wts, (int)E, (int)E, (int)E, (int)SK,
      SQ * E, SK * E, SQ * SK, 0.125f);

  // 5) softmax over each of the B*SQ rows of length SK
  softmax2048_kernel<<<(unsigned)(B * SQ), blk, 0, stream>>>(wts);

  // 6) context = weights @ WV: per batch [2048,2048] x [2048,1024] -> f32
  dim3 gv((unsigned)(E / 128), (unsigned)(SQ / 128), (unsigned)B);
  gemm_wmma_kernel<0, 2, 1, 0><<<gv, blk, 0, stream>>>(
      wts, wv, ctx, (int)SK, (int)SK, (int)E, (int)E,
      SQ * SK, SK * E, SQ * E, 1.0f);
}
